// DySAT_15367392985685
// MI455X (gfx1250) — matrix operand order repaired and checked by reading, not compiled, not run on hardware
//
#include <hip/hip_runtime.h>
#include <hip/hip_bf16.h>

#define NN 50000
#define TT 8
#define EE 500000
#define FIN 128
#define DD 64
#define HD 16
#define EPSN 1e-5f

typedef __attribute__((ext_vector_type(16))) __bf16 v16bf;
typedef __attribute__((ext_vector_type(8)))  __bf16 v8bf;
typedef __attribute__((ext_vector_type(8)))  float  v8f;

// ---- CDNA5 WMMA fragment helpers (wave32, 16x16x32 bf16) -------------------
// A (16x32, 16-bit) VGPR layout: lane m=l&15, h=l>>4:
//   VGPR0..3 = K 8h..8h+7, VGPR4..7 = K 16+8h..16+8h+7  (two contiguous 16B runs)
__device__ __forceinline__ v16bf ldsA_rm(const __bf16* base, int ld, int lane) {
  int m = lane & 15, h = lane >> 4;
  const __bf16* p = base + m * ld + (h << 3);
  v8bf lo = *(const v8bf*)(p);
  v8bf hi = *(const v8bf*)(p + 16);
  return __builtin_shufflevector(lo, hi, 0, 1, 2, 3, 4, 5, 6, 7,
                                 8, 9, 10, 11, 12, 13, 14, 15);
}

// B fragment from pre-swizzled tile: lane's 16 elements contiguous (32B)
__device__ __forceinline__ v16bf ldsB_swz(const __bf16* tbase, int lane) {
  const __bf16* p = tbase + (lane << 4);
  v8bf lo = *(const v8bf*)(p);
  v8bf hi = *(const v8bf*)(p + 8);
  return __builtin_shufflevector(lo, hi, 0, 1, 2, 3, 4, 5, 6, 7,
                                 8, 9, 10, 11, 12, 13, 14, 15);
}

__device__ __forceinline__ v8f wmma_bf16(v16bf a, v16bf b, v8f c) {
  return __builtin_amdgcn_wmma_f32_16x16x32_bf16(false, a, false, b, (short)0, c,
                                                 false, false);
}

__device__ __forceinline__ v8f vz() {
  v8f z = {0.f, 0.f, 0.f, 0.f, 0.f, 0.f, 0.f, 0.f};
  return z;
}

// Swizzle a KxN fp32 weight into fragment-ordered bf16 tiles:
// tile (kt,ct) at offset (kt*(N/16)+ct)*512; within tile lane*16 + e, where
// element e of lane l is W[(kt*32 + 8*(l>>4) + e + (e>=8 ? 8 : 0))*N + ct*16 + (l&15)].
__device__ __forceinline__ void load_swz(const float* __restrict__ W,
                                         __bf16* __restrict__ sW,
                                         int K, int N, int tid) {
  int total = K * N;
  int ntn = N >> 4;
  for (int i = tid; i < total; i += 256) {
    int tile = i >> 9, r = i & 511;
    int lane = r >> 4, e = r & 15;
    int kt = tile / ntn, ct = tile - kt * ntn;
    int k = kt * 32 + ((lane >> 4) << 3) + e + ((e >> 3) << 3);
    int n = ct * 16 + (lane & 15);
    sW[i] = (__bf16)W[k * N + n];
  }
}

// ---- Stage 1: xw = x @ Wl + bl  ([N,128] @ [128,64]) -----------------------
__global__ __launch_bounds__(256) void xw_kernel(const float* __restrict__ x,
                                                 const float* __restrict__ Wl,
                                                 const float* __restrict__ bl,
                                                 float* __restrict__ xw) {
  __shared__ __align__(16) __bf16 sX[128 * FIN];     // row-major bf16
  __shared__ __align__(16) __bf16 sWl[FIN * DD];     // swizzled tiles
  __shared__ float sbl[DD];
  int tid = threadIdx.x;
  int r0 = blockIdx.x * 128;
  for (int i = tid; i < 128 * FIN; i += 256) {
    int r = i >> 7, c = i & 127, gr = r0 + r;
    sX[i] = (__bf16)((gr < NN) ? x[(size_t)gr * FIN + c] : 0.0f);
  }
  load_swz(Wl, sWl, FIN, DD, tid);
  if (tid < DD) sbl[tid] = bl[tid];
  __syncthreads();

  int wave = tid >> 5, lane = tid & 31;
  int mrow = wave * 16;
  v8f acc[4];
#pragma unroll
  for (int i = 0; i < 4; ++i) acc[i] = vz();
#pragma unroll
  for (int kt = 0; kt < 4; ++kt) {
    v16bf a = ldsA_rm(&sX[mrow * FIN + kt * 32], FIN, lane);
#pragma unroll
    for (int ct = 0; ct < 4; ++ct) {
      v16bf b = ldsB_swz(&sWl[(kt * 4 + ct) << 9], lane);
      acc[ct] = wmma_bf16(a, b, acc[ct]);
    }
  }
  int hhalf = lane >> 4, nn = lane & 15;
#pragma unroll
  for (int ct = 0; ct < 4; ++ct)
#pragma unroll
    for (int r = 0; r < 8; ++r) {
      int row = mrow + r + 8 * hhalf;
      int gr = r0 + row, col = ct * 16 + nn;
      if (gr < NN) xw[(size_t)gr * DD + col] = acc[ct][r] + sbl[col];
    }
}

// ---- Stage 2: Z[dst,t,:] += w * xw[src,:]  (one wave per edge) -------------
__global__ __launch_bounds__(256) void scatter_kernel(const float* __restrict__ xw,
                                                      const int* __restrict__ src,
                                                      const int* __restrict__ dst,
                                                      const float* __restrict__ ew,
                                                      float* __restrict__ Z) {
  const long long total = (long long)TT * EE;
  int lane = threadIdx.x & 31;
  long long wid = (long long)blockIdx.x * (blockDim.x >> 5) + (threadIdx.x >> 5);
  long long nw = (long long)gridDim.x * (blockDim.x >> 5);
  for (long long e = wid; e < total; e += nw) {
    int t = (int)(e / EE);
    int s = src[e], d = dst[e];
    float w = ew[e];
    float v0 = xw[(size_t)s * DD + lane] * w;
    float v1 = xw[(size_t)s * DD + 32 + lane] * w;
    float* zp = &Z[((size_t)d * TT + t) * DD];
    unsafeAtomicAdd(&zp[lane], v0);
    unsafeAtomicAdd(&zp[32 + lane], v1);
  }
}

// ---- Stage 3: fused temporal attention + FFN (16 nodes / block) ------------
__global__ __launch_bounds__(256) void temporal_kernel(
    const float* __restrict__ Zg,
    const float* __restrict__ Wq, const float* __restrict__ bq,
    const float* __restrict__ Wk, const float* __restrict__ bk,
    const float* __restrict__ Wv, const float* __restrict__ bv,
    const float* __restrict__ Wo, const float* __restrict__ bo,
    const float* __restrict__ W1, const float* __restrict__ b1,
    const float* __restrict__ W2, const float* __restrict__ b2,
    const float* __restrict__ g1, const float* __restrict__ be1,
    const float* __restrict__ g2, const float* __restrict__ be2,
    float* __restrict__ out) {
  __shared__ __align__(16) __bf16 sWq[DD * DD], sWk[DD * DD],
                                  sWv[DD * DD], sWo[DD * DD];      // swizzled
  __shared__ __align__(16) __bf16 sW1[DD * 2 * DD], sW2[2 * DD * DD];
  __shared__ float sB[704];            // bq bk bv bo | b1(128) | b2 | g1 be1 g2 be2
  __shared__ __align__(16) float sZ[128 * DD];      // relu(Z), residual 1
  __shared__ __align__(16) __bf16 sZbf[128 * DD];   // bf16 shadow for A-frags
  __shared__ __align__(16) float sP[3 * 128 * DD];  // Q|K|V phase scratch (96KB)

  int tid = threadIdx.x;
  load_swz(Wq, sWq, DD, DD, tid);
  load_swz(Wk, sWk, DD, DD, tid);
  load_swz(Wv, sWv, DD, DD, tid);
  load_swz(Wo, sWo, DD, DD, tid);
  load_swz(W1, sW1, DD, 2 * DD, tid);
  load_swz(W2, sW2, 2 * DD, DD, tid);
  if (tid < 64) {
    sB[tid] = bq[tid];        sB[64 + tid] = bk[tid];
    sB[128 + tid] = bv[tid];  sB[192 + tid] = bo[tid];
    sB[256 + tid] = b1[tid];  sB[320 + tid] = b1[64 + tid];
    sB[384 + tid] = b2[tid];
    sB[448 + tid] = g1[tid];  sB[512 + tid] = be1[tid];
    sB[576 + tid] = g2[tid];  sB[640 + tid] = be2[tid];
  }
  size_t gbase = (size_t)blockIdx.x * 128 * DD;  // 16 nodes * T * D, contiguous
  for (int i = tid; i < (128 * DD) / 4; i += 256) {
    float4 v = ((const float4*)(Zg + gbase))[i];
    v.x = fmaxf(v.x, 0.f); v.y = fmaxf(v.y, 0.f);
    v.z = fmaxf(v.z, 0.f); v.w = fmaxf(v.w, 0.f);
    ((float4*)sZ)[i] = v;
    __bf16* zb = &sZbf[i * 4];
    zb[0] = (__bf16)v.x; zb[1] = (__bf16)v.y;
    zb[2] = (__bf16)v.z; zb[3] = (__bf16)v.w;
  }
  __syncthreads();

  int wave = tid >> 5, lane = tid & 31;
  int mrow = wave * 16;            // 16 rows = 2 nodes * T per wave
  int hhalf = lane >> 4, nn = lane & 15;
  float* sQ = sP;
  float* sK = sP + 8192;
  float* sV = sP + 16384;

  // ---- Q/K/V projections ([16,64] @ [64,64] per wave) ----
  const __bf16* Wm[3] = {sWq, sWk, sWv};
  float* Om[3] = {sQ, sK, sV};
  const int bo3[3] = {0, 64, 128};
#pragma unroll
  for (int p = 0; p < 3; ++p) {
    v8f acc[4];
#pragma unroll
    for (int i = 0; i < 4; ++i) acc[i] = vz();
#pragma unroll
    for (int kt = 0; kt < 2; ++kt) {
      v16bf a = ldsA_rm(&sZbf[mrow * DD + kt * 32], DD, lane);
#pragma unroll
      for (int ct = 0; ct < 4; ++ct) {
        v16bf b = ldsB_swz(&Wm[p][(kt * 4 + ct) << 9], lane);
        acc[ct] = wmma_bf16(a, b, acc[ct]);
      }
    }
#pragma unroll
    for (int ct = 0; ct < 4; ++ct)
#pragma unroll
      for (int r = 0; r < 8; ++r) {
        int row = mrow + r + 8 * hhalf, col = ct * 16 + nn;
        Om[p][row * DD + col] = acc[ct][r] + sB[bo3[p] + col];
      }
  }

  // ---- per-node 4-head attention over T=8 (2 (node,h,t) combos per lane) ----
  float aout[2][HD];
#pragma unroll
  for (int rep = 0; rep < 2; ++rep) {
    int combo = lane + 32 * rep;
    int nl = combo >> 5, h = (combo >> 3) & 3, t = combo & 7;
    int base = mrow + nl * 8;
    const float* qr = &sQ[(base + t) * DD + h * HD];
    float sc[TT];
    float mx = -1e30f;
#pragma unroll
    for (int s = 0; s < TT; ++s) {
      const float* kr = &sK[(base + s) * DD + h * HD];
      float d = 0.f;
#pragma unroll
      for (int e = 0; e < HD; ++e) d += qr[e] * kr[e];
      sc[s] = d * 0.25f;  // 1/sqrt(16)
      mx = fmaxf(mx, sc[s]);
    }
    float sum = 0.f;
#pragma unroll
    for (int s = 0; s < TT; ++s) { sc[s] = __expf(sc[s] - mx); sum += sc[s]; }
    float inv = 1.0f / sum;
#pragma unroll
    for (int e = 0; e < HD; ++e) {
      float o = 0.f;
#pragma unroll
      for (int s = 0; s < TT; ++s) o += sc[s] * sV[(base + s) * DD + h * HD + e];
      aout[rep][e] = o * inv;
    }
  }
  // all Q/K/V reads complete (in-order DS per wave); attn (bf16) -> Q slot
  __bf16* sAttnBf = (__bf16*)sQ;  // 128x64 bf16 (16KB of 32KB slot)
#pragma unroll
  for (int rep = 0; rep < 2; ++rep) {
    int combo = lane + 32 * rep;
    int nl = combo >> 5, h = (combo >> 3) & 3, t = combo & 7;
    int base = mrow + nl * 8;
    __bf16* dp = &sAttnBf[(base + t) * DD + h * HD];
    v8bf p0, p1;
#pragma unroll
    for (int e = 0; e < 8; ++e) p0[e] = (__bf16)aout[rep][e];
#pragma unroll
    for (int e = 0; e < 8; ++e) p1[e] = (__bf16)aout[rep][8 + e];
    *(v8bf*)(dp) = p0;
    *(v8bf*)(dp + 8) = p1;
  }

  // ---- attn @ Wo + bo + Z -> h1 (pre-LN), K slot ----
  float* sH1 = sK;
  {
    v8f acc[4];
#pragma unroll
    for (int i = 0; i < 4; ++i) acc[i] = vz();
#pragma unroll
    for (int kt = 0; kt < 2; ++kt) {
      v16bf a = ldsA_rm(&sAttnBf[mrow * DD + kt * 32], DD, lane);
#pragma unroll
      for (int ct = 0; ct < 4; ++ct) {
        v16bf b = ldsB_swz(&sWo[(kt * 4 + ct) << 9], lane);
        acc[ct] = wmma_bf16(a, b, acc[ct]);
      }
    }
#pragma unroll
    for (int ct = 0; ct < 4; ++ct)
#pragma unroll
      for (int r = 0; r < 8; ++r) {
        int row = mrow + r + 8 * hhalf, col = ct * 16 + nn;
        sH1[row * DD + col] = acc[ct][r] + sB[192 + col] + sZ[row * DD + col];
      }
  }
  // LN1 in place (lane < 16 owns a row) + bf16 shadow into V slot
  __bf16* sH1bf = (__bf16*)sV;
  if (lane < 16) {
    int row = mrow + lane;
    float* rp = &sH1[row * DD];
    float m = 0.f;
#pragma unroll
    for (int c = 0; c < DD; ++c) m += rp[c];
    m *= (1.0f / DD);
    float v = 0.f;
#pragma unroll
    for (int c = 0; c < DD; ++c) { float d = rp[c] - m; v += d * d; }
    float is = rsqrtf(v * (1.0f / DD) + EPSN);
#pragma unroll
    for (int c0 = 0; c0 < DD; c0 += 8) {
      v8bf pk;
#pragma unroll
      for (int j = 0; j < 8; ++j) {
        float nv = (rp[c0 + j] - m) * is * sB[448 + c0 + j] + sB[512 + c0 + j];
        rp[c0 + j] = nv;
        pk[j] = (__bf16)nv;
      }
      *(v8bf*)(&sH1bf[row * DD + c0]) = pk;
    }
  }

  // ---- FFN: relu(h1 @ W1 + b1) -> mid (bf16, Q slot: 128x128 = 32KB) ----
  __bf16* sMidBf = (__bf16*)sQ;
  {
    v8f acc[8];
#pragma unroll
    for (int i = 0; i < 8; ++i) acc[i] = vz();
#pragma unroll
    for (int kt = 0; kt < 2; ++kt) {
      v16bf a = ldsA_rm(&sH1bf[mrow * DD + kt * 32], DD, lane);
#pragma unroll
      for (int ct = 0; ct < 8; ++ct) {
        v16bf b = ldsB_swz(&sW1[(kt * 8 + ct) << 9], lane);
        acc[ct] = wmma_bf16(a, b, acc[ct]);
      }
    }
#pragma unroll
    for (int ct = 0; ct < 8; ++ct)
#pragma unroll
      for (int r = 0; r < 8; ++r) {
        int row = mrow + r + 8 * hhalf, col = ct * 16 + nn;
        sMidBf[row * 128 + col] = (__bf16)fmaxf(acc[ct][r] + sB[256 + col], 0.0f);
      }
  }

  // ---- mid @ W2 + b2 + h1 -> h2 (in place over sH1) ----
  {
    v8f acc[4];
#pragma unroll
    for (int i = 0; i < 4; ++i) acc[i] = vz();
#pragma unroll
    for (int kt = 0; kt < 4; ++kt) {
      v16bf a = ldsA_rm(&sMidBf[mrow * 128 + kt * 32], 128, lane);
#pragma unroll
      for (int ct = 0; ct < 4; ++ct) {
        v16bf b = ldsB_swz(&sW2[(kt * 4 + ct) << 9], lane);
        acc[ct] = wmma_bf16(a, b, acc[ct]);
      }
    }
#pragma unroll
    for (int ct = 0; ct < 4; ++ct)
#pragma unroll
      for (int r = 0; r < 8; ++r) {
        int row = mrow + r + 8 * hhalf, col = ct * 16 + nn;
        sH1[row * DD + col] += acc[ct][r] + sB[384 + col];
      }
  }
  // LN2 + vectorized store
  if (lane < 16) {
    int row = mrow + lane;
    float* rp = &sH1[row * DD];
    float m = 0.f;
#pragma unroll
    for (int c = 0; c < DD; ++c) m += rp[c];
    m *= (1.0f / DD);
    float v = 0.f;
#pragma unroll
    for (int c = 0; c < DD; ++c) { float d = rp[c] - m; v += d * d; }
    float is = rsqrtf(v * (1.0f / DD) + EPSN);
    float* op = &out[gbase + (size_t)row * DD];
#pragma unroll
    for (int c0 = 0; c0 < DD; c0 += 4) {
      float4 o;
      o.x = (rp[c0 + 0] - m) * is * sB[576 + c0 + 0] + sB[640 + c0 + 0];
      o.y = (rp[c0 + 1] - m) * is * sB[576 + c0 + 1] + sB[640 + c0 + 1];
      o.z = (rp[c0 + 2] - m) * is * sB[576 + c0 + 2] + sB[640 + c0 + 2];
      o.w = (rp[c0 + 3] - m) * is * sB[576 + c0 + 3] + sB[640 + c0 + 3];
      *(float4*)(op + c0) = o;
    }
  }
}

// ---------------------------------------------------------------------------
extern "C" void kernel_launch(void* const* d_in, const int* in_sizes, int n_in,
                              void* d_out, int out_size, void* d_ws, size_t ws_size,
                              hipStream_t stream) {
  (void)in_sizes; (void)n_in; (void)out_size; (void)ws_size;
  const float* x    = (const float*)d_in[0];
  const int*   esrc = (const int*)d_in[1];
  const int*   edst = (const int*)d_in[2];
  const float* ew   = (const float*)d_in[3];
  const float* Wl   = (const float*)d_in[4];
  const float* bl   = (const float*)d_in[5];
  const float* Wq   = (const float*)d_in[6];
  const float* bq   = (const float*)d_in[7];
  const float* Wk   = (const float*)d_in[8];
  const float* bk   = (const float*)d_in[9];
  const float* Wv   = (const float*)d_in[10];
  const float* bv   = (const float*)d_in[11];
  const float* Wo   = (const float*)d_in[12];
  const float* bo   = (const float*)d_in[13];
  const float* W1   = (const float*)d_in[14];
  const float* b1   = (const float*)d_in[15];
  const float* W2   = (const float*)d_in[16];
  const float* b2   = (const float*)d_in[17];
  const float* g1   = (const float*)d_in[18];
  const float* be1  = (const float*)d_in[19];
  const float* g2   = (const float*)d_in[20];
  const float* be2  = (const float*)d_in[21];
  float* out = (float*)d_out;

  float* xw = (float*)d_ws;                 // N*D floats
  float* Z  = xw + (size_t)NN * DD;         // N*T*D floats

  xw_kernel<<<(NN + 127) / 128, 256, 0, stream>>>(x, Wl, bl, xw);
  hipMemsetAsync(Z, 0, (size_t)NN * TT * DD * sizeof(float), stream);
  scatter_kernel<<<8192, 256, 0, stream>>>(xw, esrc, edst, ew, Z);
  temporal_kernel<<<NN / 16, 256, 0, stream>>>(Z, Wq, bq, Wk, bk, Wv, bv, Wo, bo,
                                               W1, b1, W2, b2, g1, be1, g2, be2, out);
}
